// MultiHeadAttention_46969762349562
// MI455X (gfx1250) — matrix operand order repaired and checked by reading, not compiled, not run on hardware
//
#include <hip/hip_runtime.h>

// ---------------------------------------------------------------------------
// MHA forward for gfx1250 (MI455X): bf16 WMMA GEMMs + flash attention.
// Pipeline: cvt(fp32->bf16) -> 3x proj GEMM -> flash attn -> output GEMM.
// LDS staging uses GLOBAL_LOAD_ASYNC_TO_LDS (ASYNCcnt) when available.
// ---------------------------------------------------------------------------

typedef __bf16 bf16;
typedef __attribute__((ext_vector_type(16))) __bf16 v16bf;
typedef __attribute__((ext_vector_type(8)))  __bf16 v8bf;
typedef __attribute__((ext_vector_type(8)))  float  v8f;
typedef __attribute__((ext_vector_type(4)))  int    v4i;

union Frag {
    v16bf v;
    v8bf  h[2];
    bf16  e[16];
};

constexpr int BB = 2;
constexpr int SS = 2048;
constexpr int DM = 1024;
constexpr int HH = 16;
constexpr int DH = 64;
constexpr int MM = BB * SS;   // 4096 rows

// ---- CDNA5 async global->LDS copy (ASYNCcnt) --------------------------------
#if defined(__has_builtin)
#if __has_builtin(__builtin_amdgcn_global_load_async_to_lds_b128) && \
    __has_builtin(__builtin_amdgcn_s_wait_asynccnt)
#define HAS_ASYNC_LDS 1
#endif
#endif

__device__ __forceinline__ void async_copy_b128(const bf16* g, bf16* l) {
#ifdef HAS_ASYNC_LDS
    __builtin_amdgcn_global_load_async_to_lds_b128(
        (__attribute__((address_space(1))) v4i*)(g),
        (__attribute__((address_space(3))) v4i*)(l),
        0, 0);
#else
    *(v8bf*)l = *(const v8bf*)g;
#endif
}

__device__ __forceinline__ void async_wait() {
#ifdef HAS_ASYNC_LDS
    __builtin_amdgcn_s_wait_asynccnt(0);
#endif
}

// ---------------------------------------------------------------------------
// fp32 -> bf16 conversion
// ---------------------------------------------------------------------------
__global__ void __launch_bounds__(256) cvt_f32_bf16(const float* __restrict__ src,
                                                    bf16* __restrict__ dst, int n) {
    int i = blockIdx.x * 256 + threadIdx.x;
    if (i < n) dst[i] = (bf16)src[i];
}

// ---------------------------------------------------------------------------
// NT GEMM: Out[m,n] = sum_k A[m,k] * W[n,k] + bias[n]
// A: [4096,1024] bf16 row-major, W: [1024,1024] bf16 row-major.
// proj==1: write bf16 to [B,H,S,Dh];  proj==0: write fp32 flat [4096,1024].
// Block tile 128(M) x 128(N), 8 waves (2x4 grid), wave tile 64x32, K-step 32.
// Double-buffered LDS (2 x 16KB) with async global->LDS staging.
// ---------------------------------------------------------------------------
__global__ void __launch_bounds__(256) gemm_bf16_nt(
    const bf16* __restrict__ A, const bf16* __restrict__ W,
    const float* __restrict__ bias,
    bf16* __restrict__ outProj, float* __restrict__ outFlat, int proj) {

    __shared__ alignas(16) bf16 sA[2][128][32];
    __shared__ alignas(16) bf16 sB[2][128][32];

    const int t    = threadIdx.x;
    const int wave = t >> 5;
    const int lane = t & 31;
    const int lh   = lane >> 4;      // half-wave select
    const int ln   = lane & 15;
    const int wm   = wave >> 2;      // wave M index (0..1), 64 rows each
    const int wn   = wave & 3;       // wave N index (0..3), 32 cols each
    const int m0   = blockIdx.y * 128;
    const int n0   = blockIdx.x * 128;

    // staging coordinates: 512 x 16B chunks per 128x32 tile, 2 per thread
    const int sr0 = (t + 0)   >> 2, sc0 = ((t + 0)   & 3) * 8;
    const int sr1 = (t + 256) >> 2, sc1 = ((t + 256) & 3) * 8;

    v8f c[4][2] = {};

    // prologue: stage K-tile 0 into buffer 0
    async_copy_b128(A + (size_t)(m0 + sr0) * DM + sc0, &sA[0][sr0][sc0]);
    async_copy_b128(A + (size_t)(m0 + sr1) * DM + sc1, &sA[0][sr1][sc1]);
    async_copy_b128(W + (size_t)(n0 + sr0) * DM + sc0, &sB[0][sr0][sc0]);
    async_copy_b128(W + (size_t)(n0 + sr1) * DM + sc1, &sB[0][sr1][sc1]);
    async_wait();
    __syncthreads();

    for (int k0 = 0; k0 < DM; k0 += 32) {
        const int cur = (k0 >> 5) & 1;

        // stage next K-tile into the other buffer while we compute
        if (k0 + 32 < DM) {
            const int nxt = cur ^ 1;
            const int kn  = k0 + 32;
            async_copy_b128(A + (size_t)(m0 + sr0) * DM + kn + sc0, &sA[nxt][sr0][sc0]);
            async_copy_b128(A + (size_t)(m0 + sr1) * DM + kn + sc1, &sA[nxt][sr1][sc1]);
            async_copy_b128(W + (size_t)(n0 + sr0) * DM + kn + sc0, &sB[nxt][sr0][sc0]);
            async_copy_b128(W + (size_t)(n0 + sr1) * DM + kn + sc1, &sB[nxt][sr1][sc1]);
        }

        // ---- fragments (contiguous K order k = 16*lh + s, same for A and B)
        Frag a[4], b[2];
        #pragma unroll
        for (int im = 0; im < 4; ++im) {
            int r = wm * 64 + im * 16 + ln;   // A: lane = M row
            a[im].h[0] = *(v8bf*)&sA[cur][r][16 * lh];      // K run 0..15/16..31
            a[im].h[1] = *(v8bf*)&sA[cur][r][16 * lh + 8];
        }
        #pragma unroll
        for (int in = 0; in < 2; ++in) {
            int r = wn * 32 + in * 16 + ln;   // B: lane = N col (= W row)
            b[in].h[0] = *(v8bf*)&sB[cur][r][16 * lh];      // K run 0..15/16..31
            b[in].h[1] = *(v8bf*)&sB[cur][r][16 * lh + 8];
        }

        #pragma unroll
        for (int im = 0; im < 4; ++im)
            #pragma unroll
            for (int in = 0; in < 2; ++in)
                c[im][in] = __builtin_amdgcn_wmma_f32_16x16x32_bf16(
                    false, a[im].v, false, b[in].v,
                    (short)0, c[im][in], false, false);

        async_wait();        // this wave's copies into nxt landed in LDS
        __syncthreads();     // all waves' copies visible; all done reading cur
    }

    // ---- epilogue: C layout: VGPR i -> row i + 8*lh, col = ln
    #pragma unroll
    for (int im = 0; im < 4; ++im)
        #pragma unroll
        for (int in = 0; in < 2; ++in)
            #pragma unroll
            for (int i = 0; i < 8; ++i) {
                int r   = m0 + wm * 64 + im * 16 + i + 8 * lh;
                int col = n0 + wn * 32 + in * 16 + ln;
                float val = c[im][in][i] + bias[col];
                if (proj) {
                    int bb = r >> 11, s = r & 2047;
                    int hh = col >> 6, d = col & 63;
                    outProj[(((size_t)bb * HH + hh) * SS + s) * DH + d] = (bf16)val;
                } else {
                    outFlat[(size_t)r * DM + col] = val;
                }
            }
}

// ---------------------------------------------------------------------------
// Flash attention: one wave per 16 query rows per (b,h).
// Q,K,V in [B,H,S,Dh] bf16. Streams 32 keys/iter with online softmax.
// Writes attention output as bf16 [B,S,D] (GEMM-input layout for Wo).
// ---------------------------------------------------------------------------
__global__ void __launch_bounds__(32) flash_attn(
    const bf16* __restrict__ Qp, const bf16* __restrict__ Kp,
    const bf16* __restrict__ Vp, bf16* __restrict__ Xa) {

    __shared__ alignas(16) bf16 sP[16][32];
    __shared__ alignas(16) bf16 sV[32][64];

    const int lane = threadIdx.x;
    const int lh   = lane >> 4;
    const int ln   = lane & 15;
    const int off  = 16 * lh;
    const int q0   = blockIdx.x * 16;
    const int h    = blockIdx.y;
    const int b    = blockIdx.z;

    const size_t hb = ((size_t)b * HH + h) * (size_t)SS * DH;
    const bf16* Qh = Qp + hb;
    const bf16* Kh = Kp + hb;
    const bf16* Vh = Vp + hb;

    // Q fragments (A layout), Dh = 64 -> two K-chunks of 32
    Frag qa[2];
    #pragma unroll
    for (int cdx = 0; cdx < 2; ++cdx) {
        const bf16* p = Qh + (size_t)(q0 + ln) * DH + cdx * 32 + off;
        qa[cdx].h[0] = *(const v8bf*)p;
        qa[cdx].h[1] = *(const v8bf*)(p + 8);
    }

    v8f o[4] = {};
    float mi[8], li[8];
    #pragma unroll
    for (int i = 0; i < 8; ++i) { mi[i] = -1e30f; li[i] = 0.0f; }

    for (int j = 0; j < SS; j += 32) {
        // ---- stage V tile (32 keys x 64) to LDS (async; overlaps QK^T)
        #pragma unroll
        for (int u = 0; u < 8; ++u)
            async_copy_b128(Vh + (size_t)(j + lane) * DH + u * 8, &sV[lane][u * 8]);

        // ---- scores: two 16x16 tiles over 32 keys
        v8f s[2];
        #pragma unroll
        for (int tt = 0; tt < 2; ++tt) {
            Frag kb[2];
            #pragma unroll
            for (int cdx = 0; cdx < 2; ++cdx) {
                const bf16* kp =
                    Kh + (size_t)(j + tt * 16 + ln) * DH + cdx * 32 + 16 * lh;
                kb[cdx].h[0] = *(const v8bf*)kp;
                kb[cdx].h[1] = *(const v8bf*)(kp + 8);
            }
            v8f acc = {};
            acc = __builtin_amdgcn_wmma_f32_16x16x32_bf16(
                false, qa[0].v, false, kb[0].v, (short)0, acc, false, false);
            acc = __builtin_amdgcn_wmma_f32_16x16x32_bf16(
                false, qa[1].v, false, kb[1].v, (short)0, acc, false, false);
            s[tt] = acc;
        }
        #pragma unroll
        for (int i = 0; i < 8; ++i) { s[0][i] *= 0.125f; s[1][i] *= 0.125f; }

        // ---- online softmax per row (rows striped: VGPR i -> row i + 8*lh)
        #pragma unroll
        for (int i = 0; i < 8; ++i) {
            float p0 = s[0][i], p1 = s[1][i];
            float mx = fmaxf(p0, p1);
            mx = fmaxf(mx, __shfl_xor(mx, 1));
            mx = fmaxf(mx, __shfl_xor(mx, 2));
            mx = fmaxf(mx, __shfl_xor(mx, 4));
            mx = fmaxf(mx, __shfl_xor(mx, 8));
            float mnew  = fmaxf(mi[i], mx);
            float alpha = __expf(mi[i] - mnew);
            p0 = __expf(p0 - mnew);
            p1 = __expf(p1 - mnew);
            float rs = p0 + p1;
            rs += __shfl_xor(rs, 1);
            rs += __shfl_xor(rs, 2);
            rs += __shfl_xor(rs, 4);
            rs += __shfl_xor(rs, 8);
            li[i] = li[i] * alpha + rs;
            mi[i] = mnew;
            #pragma unroll
            for (int tt = 0; tt < 4; ++tt) o[tt][i] *= alpha;
            int r = i + 8 * lh;
            sP[r][ln]      = (bf16)p0;
            sP[r][16 + ln] = (bf16)p1;
        }
        async_wait();        // V tile landed in LDS
        __syncthreads();

        // ---- P as A fragment (16x32)
        Frag pa;
        pa.h[0] = *(v8bf*)&sP[ln][off];
    pa.h[1] = *(v8bf*)&sP[ln][off + 8];

        // ---- V as B fragments: lane = d column, K = key index (16-run/half)
        #pragma unroll
        for (int tt = 0; tt < 4; ++tt) {
            Frag vb;
            int d = tt * 16 + ln;
            #pragma unroll
            for (int kk = 0; kk < 16; ++kk)
                vb.e[kk] = sV[16 * lh + kk][d];
            o[tt] = __builtin_amdgcn_wmma_f32_16x16x32_bf16(
                false, pa.v, false, vb.v, (short)0, o[tt], false, false);
        }
        __syncthreads();
    }

    // ---- normalize and write out as [B,S,D] bf16
    #pragma unroll
    for (int i = 0; i < 8; ++i) {
        float inv = 1.0f / li[i];
        int r = q0 + i + 8 * lh;
        #pragma unroll
        for (int tt = 0; tt < 4; ++tt) {
            int col = h * DH + tt * 16 + ln;
            Xa[((size_t)b * SS + r) * DM + col] = (bf16)(o[tt][i] * inv);
        }
    }
}

// ---------------------------------------------------------------------------
// launch
// ---------------------------------------------------------------------------
extern "C" void kernel_launch(void* const* d_in, const int* in_sizes, int n_in,
                              void* d_out, int out_size, void* d_ws, size_t ws_size,
                              hipStream_t stream) {
    (void)in_sizes; (void)n_in; (void)out_size; (void)ws_size;

    const float* q  = (const float*)d_in[0];
    const float* k  = (const float*)d_in[1];
    const float* v  = (const float*)d_in[2];
    const float* Wq = (const float*)d_in[3];
    const float* bq = (const float*)d_in[4];
    const float* Wk = (const float*)d_in[5];
    const float* bk = (const float*)d_in[6];
    const float* Wv = (const float*)d_in[7];
    const float* bv = (const float*)d_in[8];
    const float* Wo = (const float*)d_in[9];
    const float* bo = (const float*)d_in[10];
    float* out = (float*)d_out;

    // workspace layout (bf16 elements); total 32M elems = 64 MB
    bf16* ws  = (bf16*)d_ws;
    const size_t ACT = (size_t)MM * DM;   // 4096*1024
    const size_t WT  = (size_t)DM * DM;   // 1024*1024
    bf16* Aq  = ws;
    bf16* Ak  = Aq  + ACT;
    bf16* Av  = Ak  + ACT;
    bf16* Wqb = Av  + ACT;
    bf16* Wkb = Wqb + WT;
    bf16* Wvb = Wkb + WT;
    bf16* Wob = Wvb + WT;
    bf16* Qp  = Wob + WT;
    bf16* Kpp = Qp  + ACT;
    bf16* Vpp = Kpp + ACT;
    bf16* Xa  = Vpp + ACT;

    const int nAct = (int)ACT, nW = (int)WT;
    cvt_f32_bf16<<<(nAct + 255) / 256, 256, 0, stream>>>(q,  Aq,  nAct);
    cvt_f32_bf16<<<(nAct + 255) / 256, 256, 0, stream>>>(k,  Ak,  nAct);
    cvt_f32_bf16<<<(nAct + 255) / 256, 256, 0, stream>>>(v,  Av,  nAct);
    cvt_f32_bf16<<<(nW   + 255) / 256, 256, 0, stream>>>(Wq, Wqb, nW);
    cvt_f32_bf16<<<(nW   + 255) / 256, 256, 0, stream>>>(Wk, Wkb, nW);
    cvt_f32_bf16<<<(nW   + 255) / 256, 256, 0, stream>>>(Wv, Wvb, nW);
    cvt_f32_bf16<<<(nW   + 255) / 256, 256, 0, stream>>>(Wo, Wob, nW);

    dim3 gGemm(DM / 128, MM / 128);   // 8 x 32 blocks
    gemm_bf16_nt<<<gGemm, 256, 0, stream>>>(Aq, Wqb, bq, Qp,  nullptr, 1);
    gemm_bf16_nt<<<gGemm, 256, 0, stream>>>(Ak, Wkb, bk, Kpp, nullptr, 1);
    gemm_bf16_nt<<<gGemm, 256, 0, stream>>>(Av, Wvb, bv, Vpp, nullptr, 1);

    dim3 gAttn(SS / 16, HH, BB);     // 128 x 16 x 2
    flash_attn<<<gAttn, 32, 0, stream>>>(Qp, Kpp, Vpp, Xa);

    gemm_bf16_nt<<<gGemm, 256, 0, stream>>>(Xa, Wob, bo, nullptr, out, 0);
}
